// NCE_40948218200545
// MI455X (gfx1250) — matrix operand order repaired and checked by reading, not compiled, not run on hardware
//
#include <hip/hip_runtime.h>
#include <math.h>

// Problem constants (from reference)
#define N_ROWS 8192
#define M_COLS 8192
#define D_DIM  512
#define INV_T  2.0f          // 1 / TEMPERATURE (T = 0.5)
#define NEG_W  1.0f

// Tiling
#define MSPLIT 8                     // m-dimension split across blocks
#define MRANGE (M_COLS / MSPLIT)     // 1024 columns per block
#define BN 64                        // n-rows per block (4 wave-rows of 16)
#define BM 64                        // m-cols per iteration (2 wave-cols of 32)
#define KC 128                       // K chunk for B staging
#define LDKA 520                     // A plane stride (bf16 elems): 260 dwords % 64 == 4
#define LDKB 136                     // B plane stride (bf16 elems):  68 dwords % 64 == 4

typedef __attribute__((ext_vector_type(2)))  float  v2f;
typedef __attribute__((ext_vector_type(4)))  float  v4f;
typedef __attribute__((ext_vector_type(8)))  float  v8f;
typedef __attribute__((ext_vector_type(4)))  __bf16 v4bf;
typedef __attribute__((ext_vector_type(8)))  __bf16 v8bf;
typedef __attribute__((ext_vector_type(16))) __bf16 v16bf;

__device__ __forceinline__ float log_sigmoid(float z) {
  // numerically stable: min(z,0) - log1p(exp(-|z|))
  return fminf(z, 0.0f) - log1pf(__expf(-fabsf(z)));
}

__device__ __forceinline__ v16bf cat8(v8bf a, v8bf b) {
  return __builtin_shufflevector(a, b, 0, 1, 2, 3, 4, 5, 6, 7,
                                 8, 9, 10, 11, 12, 13, 14, 15);
}

// Split a float4 into bf16 hi/lo planes: x = hi + lo (+ O(2^-18) rel err)
__device__ __forceinline__ void split4(v4f v, v4bf& h, v4bf& l) {
#pragma unroll
  for (int j = 0; j < 4; ++j) {
    __bf16 hb = (__bf16)v[j];           // RNE
    h[j] = hb;
    l[j] = (__bf16)(v[j] - (float)hb);  // residual
  }
}

// ---------------------------------------------------------------------------
// Kernel A: align[n] = log_sigmoid(dot(ref[n], pos[n]) / T), one wave per row
// ---------------------------------------------------------------------------
__global__ void nce_align_kernel(const float* __restrict__ ref,
                                 const float* __restrict__ pos,
                                 float* __restrict__ out) {
  const int wave = threadIdx.x >> 5;
  const int lane = threadIdx.x & 31;
  const int n = blockIdx.x * 8 + wave;
  const float* r = ref + (size_t)n * D_DIM;
  const float* p = pos + (size_t)n * D_DIM;
  float acc = 0.0f;
#pragma unroll
  for (int i = 0; i < 4; ++i) {
    v4f a = *(const v4f*)(r + lane * 4 + i * 128);
    v4f b = *(const v4f*)(p + lane * 4 + i * 128);
    acc += a.x * b.x + a.y * b.y + a.z * b.z + a.w * b.w;
  }
#pragma unroll
  for (int off = 16; off; off >>= 1) acc += __shfl_xor(acc, off, 32);
  if (lane == 0) out[N_ROWS + n] = log_sigmoid(acc * INV_T);
}

// ---------------------------------------------------------------------------
// Kernel B: split-bf16 GEMM (ref @ neg^T / T) via V_WMMA_F32_16X16X32_BF16,
// fused log_sigmoid(-x) + row partial reduction.
//   x*y ~= xh*yh + xh*yl + xl*yh  (fp32 accumulate) -> ~fp32 accuracy at the
//   bf16 matrix rate (K=32/instr vs K=4 for the f32 WMMA path).
// Grid: (N/BN, MSPLIT); block: 256 threads = 8 waves (4 row-tiles x 2 col-tiles,
// each wave owns a 16x32 output tile = 2 accumulators).
// A (64 x 512) is split-converted ONCE per block into LDS hi/lo planes;
// B is staged/converted per (m-tile, K-chunk). Strides padded so the per-lane
// b128 fragment reads are bank-conflict-free per half-wave.
// ---------------------------------------------------------------------------
__global__ void __launch_bounds__(256)
nce_gemm_logsig_kernel(const float* __restrict__ ref,
                       const float* __restrict__ neg,
                       float* __restrict__ part) {
  extern __shared__ __attribute__((aligned(16))) char smem[];
  __bf16* Ah = (__bf16*)smem;              // BN*LDKA
  __bf16* Al = Ah + BN * LDKA;             // BN*LDKA
  __bf16* Bh = Al + BN * LDKA;             // BM*LDKB
  __bf16* Bl = Bh + BM * LDKB;             // BM*LDKB
  float*  red = (float*)(Bl + BM * LDKB);  // BN floats

  const int tid  = threadIdx.x;
  const int lane = tid & 31;
  const int wave = tid >> 5;
  const int wr   = wave >> 1;      // 0..3 : 16-row sub-tile
  const int wc   = wave & 1;       // 0..1 : 32-col sub-tile
  const int hl   = lane >> 4;      // half-wave selector (K offset 16*hl)
  const int l16  = lane & 15;

  const int n0 = blockIdx.x * BN;
  const int m0 = blockIdx.y * MRANGE;

  if (tid < BN) red[tid] = 0.0f;

  // ---- Stage + split-convert A once: 64 rows x 512 K (8192 float4) ----
#pragma unroll
  for (int i = 0; i < 32; ++i) {
    int idx = tid + i * 256;       // 0..8191
    int row = idx >> 7;            // 128 float4 per row
    int c4  = idx & 127;
    v4f v = *(const v4f*)(ref + (size_t)(n0 + row) * D_DIM + c4 * 4);
    v4bf h, l;
    split4(v, h, l);
    *(v4bf*)(&Ah[row * LDKA + c4 * 4]) = h;
    *(v4bf*)(&Al[row * LDKA + c4 * 4]) = l;
  }

  // Fragment base offsets (contiguous k = 16*hl + s, A and B alike):
  // A 16x32: lane<16 -> K[0,16); lane>=16 -> K[16,32)
  const int aOff = (wr * 16 + l16) * LDKA + 16 * hl;
  // B 32x16: lane<16 -> K[0,16); lane>=16 -> K[16,32)
  const int bOff0 = (wc * 32 + l16) * LDKB + 16 * hl;
  const int bOff1 = bOff0 + 16 * LDKB;

  float usum[8];
#pragma unroll
  for (int i = 0; i < 8; ++i) usum[i] = 0.0f;

  for (int mt = 0; mt < MRANGE; mt += BM) {
    v8f acc0 = {};
    v8f acc1 = {};
    for (int kc = 0; kc < D_DIM; kc += KC) {
      __syncthreads();
      // ---- Stage + split-convert B chunk: 64 cols x 128 K (2048 float4) ----
#pragma unroll
      for (int i = 0; i < 8; ++i) {
        int idx = tid + i * 256;   // 0..2047
        int row = idx >> 5;        // 32 float4 per row
        int c4  = idx & 31;
        v4f v = *(const v4f*)(neg + (size_t)(m0 + mt + row) * D_DIM + kc + c4 * 4);
        v4bf h, l;
        split4(v, h, l);
        *(v4bf*)(&Bh[row * LDKB + c4 * 4]) = h;
        *(v4bf*)(&Bl[row * LDKB + c4 * 4]) = l;
      }
      __syncthreads();
#pragma unroll
      for (int k = 0; k < KC; k += 32) {
        const int ai = aOff + kc + k;
        v16bf ah = cat8(*(const v8bf*)(&Ah[ai]), *(const v8bf*)(&Ah[ai + 8]));
        v16bf al = cat8(*(const v8bf*)(&Al[ai]), *(const v8bf*)(&Al[ai + 8]));
        const int b0 = bOff0 + k;
        const int b1 = bOff1 + k;
        v16bf b0h = cat8(*(const v8bf*)(&Bh[b0]), *(const v8bf*)(&Bh[b0 + 8]));
        v16bf b0l = cat8(*(const v8bf*)(&Bl[b0]), *(const v8bf*)(&Bl[b0 + 8]));
        v16bf b1h = cat8(*(const v8bf*)(&Bh[b1]), *(const v8bf*)(&Bh[b1 + 8]));
        v16bf b1l = cat8(*(const v8bf*)(&Bl[b1]), *(const v8bf*)(&Bl[b1 + 8]));
        // x*y ~= xh*yh + xh*yl + xl*yh, fp32 accumulate
        acc0 = __builtin_amdgcn_wmma_f32_16x16x32_bf16(false, ah, false, b0h,
                                                       (short)0, acc0, false, false);
        acc0 = __builtin_amdgcn_wmma_f32_16x16x32_bf16(false, ah, false, b0l,
                                                       (short)0, acc0, false, false);
        acc0 = __builtin_amdgcn_wmma_f32_16x16x32_bf16(false, al, false, b0h,
                                                       (short)0, acc0, false, false);
        acc1 = __builtin_amdgcn_wmma_f32_16x16x32_bf16(false, ah, false, b1h,
                                                       (short)0, acc1, false, false);
        acc1 = __builtin_amdgcn_wmma_f32_16x16x32_bf16(false, ah, false, b1l,
                                                       (short)0, acc1, false, false);
        acc1 = __builtin_amdgcn_wmma_f32_16x16x32_bf16(false, al, false, b1h,
                                                       (short)0, acc1, false, false);
      }
    }
    // Fused epilogue: x = dot / T ; accumulate log_sigmoid(-x) over this m-tile
#pragma unroll
    for (int i = 0; i < 8; ++i) {
      usum[i] += log_sigmoid(-(acc0[i] * INV_T));
      usum[i] += log_sigmoid(-(acc1[i] * INV_T));
    }
  }

  // Reduce across the 16 lanes holding the same output rows (tile columns).
  // C layout: VGPR i, lanes 0-15 => row i ; lanes 16-31 => row i+8.
#pragma unroll
  for (int i = 0; i < 8; ++i) {
    float v = usum[i];
    v += __shfl_xor(v, 8, 16);
    v += __shfl_xor(v, 4, 16);
    v += __shfl_xor(v, 2, 16);
    v += __shfl_xor(v, 1, 16);
    if (l16 == 0)  // exactly two contributions (wc=0,1) per row: order-independent
      atomicAdd(&red[wr * 16 + hl * 8 + i], v);
  }
  __syncthreads();
  if (tid < BN) part[(size_t)blockIdx.y * N_ROWS + n0 + tid] = red[tid];
}

// ---------------------------------------------------------------------------
// Kernel C: combine partials -> uniform, assemble the three outputs
// ---------------------------------------------------------------------------
__global__ void nce_finalize_kernel(const float* __restrict__ part,
                                    float* __restrict__ out) {
  int n = blockIdx.x * blockDim.x + threadIdx.x;
  if (n >= N_ROWS) return;
  float s = 0.0f;
#pragma unroll
  for (int i = 0; i < MSPLIT; ++i) s += part[(size_t)i * N_ROWS + n];
  float uniform = s * (1.0f / (float)M_COLS);  // REDUCE == 'mean'
  float align = out[N_ROWS + n];
  out[n] = align + NEG_W * uniform;
  out[2 * N_ROWS + n] = uniform;
}

extern "C" void kernel_launch(void* const* d_in, const int* in_sizes, int n_in,
                              void* d_out, int out_size, void* d_ws, size_t ws_size,
                              hipStream_t stream) {
  (void)in_sizes; (void)n_in; (void)out_size; (void)ws_size;
  const float* ref = (const float*)d_in[0];
  const float* pos = (const float*)d_in[1];
  const float* neg = (const float*)d_in[2];
  float* out  = (float*)d_out;
  float* part = (float*)d_ws;  // MSPLIT * N_ROWS floats = 256 KB

  nce_align_kernel<<<N_ROWS / 8, 256, 0, stream>>>(ref, pos, out);

  dim3 gridB(N_ROWS / BN, MSPLIT);
  size_t smem = (size_t)(2 * BN * LDKA + 2 * BM * LDKB) * 2 /*bytes per bf16*/
              + (size_t)BN * sizeof(float);   // ~164 KB, fits 320 KB WGP LDS
  nce_gemm_logsig_kernel<<<gridB, 256, smem, stream>>>(ref, neg, part);

  nce_finalize_kernel<<<(N_ROWS + 255) / 256, 256, 0, stream>>>(part, out);
}
